// DGCNN_35218731827840
// MI455X (gfx1250) — matrix operand order repaired and checked by reading, not compiled, not run on hardware
//
#include <hip/hip_runtime.h>

#define BATCH 8
#define NPTS  4096
#define KNN_K 20

typedef __attribute__((ext_vector_type(16))) _Float16 v16h;
typedef __attribute__((ext_vector_type(8)))  float    v8f;

enum GMode { MODE_DENSE = 0, MODE_EDGE = 1, MODE_CAT192 = 2, MODE_CAT1216 = 3 };

__device__ __forceinline__ float4 ld4(const float* p) {
  return *reinterpret_cast<const float4*>(p);
}
__device__ __forceinline__ float4 sub4(float4 a, float4 b) {
  return make_float4(a.x - b.x, a.y - b.y, a.z - b.z, a.w - b.w);
}

struct GArgs {
  const float* A;      // dense A source (MODE_DENSE)
  const float* xsrc;   // point features (MODE_EDGE)
  const int*   idx;    // knn indices    (MODE_EDGE)
  const float* x1;     // cat sources
  const float* x2;
  const float* x3;
  const float* gmax;   // [B,1024] (MODE_CAT1216)
  const float* W;      // [O, Kin] row-major
  float*       out;    // [M, O]
  int M, Kin, O;
};

// Per-row resolved pointers; segments are picked in run8().
struct RowCtx { const float* p0; const float* p1; const float* p2; const float* p3; int C; };

template<int MODE>
__device__ __forceinline__ RowCtx makeCtx(const GArgs& ga, int row) {
  RowCtx cx{};
  if (MODE == MODE_DENSE) {
    cx.p0 = ga.A + (long long)row * ga.Kin;
  } else if (MODE == MODE_EDGE) {
    int kk = row % KNN_K;
    int pn = row / KNN_K;              // b*NPTS + n
    int C  = ga.Kin >> 1;
    int b  = pn >> 12;                 // NPTS == 4096
    int nn = pn & (NPTS - 1);
    const float* xb = ga.xsrc + (long long)b * NPTS * C;
    int j = ga.idx[(long long)pn * KNN_K + kk];
    cx.p0 = xb + (long long)nn * C;    // x_i
    cx.p1 = xb + (long long)j * C;     // x_j
    cx.C  = C;
  } else if (MODE == MODE_CAT192) {
    long long r64 = (long long)row * 64;
    cx.p0 = ga.x1 + r64; cx.p1 = ga.x2 + r64; cx.p2 = ga.x3 + r64;
  } else { // MODE_CAT1216
    int b = row >> 12;
    long long r64 = (long long)row * 64;
    cx.p0 = ga.gmax + b * 1024;
    cx.p1 = ga.x1 + r64; cx.p2 = ga.x2 + r64; cx.p3 = ga.x3 + r64;
  }
  return cx;
}

// 8 consecutive f32 at 8-aligned k8; runs never cross segment boundaries
// (all boundaries are multiples of 8). Unconditional b128 loads.
template<int MODE>
__device__ __forceinline__ void run8(const RowCtx& cx, int k8, float4& lo, float4& hi) {
  if (MODE == MODE_DENSE) {
    lo = ld4(cx.p0 + k8); hi = ld4(cx.p0 + k8 + 4);
  } else if (MODE == MODE_EDGE) {
    if (k8 < cx.C) {
      float4 j0 = ld4(cx.p1 + k8), j1 = ld4(cx.p1 + k8 + 4);
      float4 i0 = ld4(cx.p0 + k8), i1 = ld4(cx.p0 + k8 + 4);
      lo = sub4(j0, i0); hi = sub4(j1, i1);
    } else {
      int q = k8 - cx.C;
      lo = ld4(cx.p0 + q); hi = ld4(cx.p0 + q + 4);
    }
  } else if (MODE == MODE_CAT192) {
    const float* p = (k8 < 64) ? (cx.p0 + k8)
                   : (k8 < 128) ? (cx.p1 + (k8 - 64))
                                : (cx.p2 + (k8 - 128));
    lo = ld4(p); hi = ld4(p + 4);
  } else { // CAT1216: [gmax 1024 | x1 64 | x2 64 | x3 64]
    const float* p;
    if (k8 < 1024)      p = cx.p0 + k8;
    else if (k8 < 1088) p = cx.p1 + (k8 - 1024);
    else if (k8 < 1152) p = cx.p2 + (k8 - 1088);
    else                p = cx.p3 + (k8 - 1152);
    lo = ld4(p); hi = ld4(p + 4);
  }
}

__device__ __forceinline__ void pack8(v16h& v, int base, float4 lo, float4 hi) {
  v[base + 0] = (_Float16)lo.x; v[base + 1] = (_Float16)lo.y;
  v[base + 2] = (_Float16)lo.z; v[base + 3] = (_Float16)lo.w;
  v[base + 4] = (_Float16)hi.x; v[base + 5] = (_Float16)hi.y;
  v[base + 6] = (_Float16)hi.z; v[base + 7] = (_Float16)hi.w;
}

// A-frag (16-bit A 16x32, ISA 7.12.2): lane l, h=l>>4, m=l&15;
//   af[j]   = A[m][kbase + 8h + j]       (j = 0..7)
//   af[8+j] = A[m][kbase + 16 + 8h + j]  (j = 0..7)
template<int MODE>
__device__ __forceinline__ v16h loadA(const RowCtx& cx, int kbase, int hh) {
  float4 a0, a1, b0, b1;
  run8<MODE>(cx, kbase + 8 * hh, a0, a1);
  run8<MODE>(cx, kbase + 16 + 8 * hh, b0, b1);
  v16h af;
  pack8(af, 0, a0, a1);
  pack8(af, 8, b0, b1);
  return af;
}

// ---------------------------------------------------------------------------
// WMMA GEMM: out[M,O] = A[M,Kin] * W[O,Kin]^T. Kin % 32 == 0, M % 64 == 0.
// One wave computes 4 vertical 16x16 tiles (64 rows x 16 cols); B-frag reused.
// B-frag (16-bit B 32x16): lane l, h=l>>4, n=l&15; bf[j] = W[o0+n][kbase+8h+j], bf[8+j] = W[o0+n][kbase+16+8h+j].
// D (f32 16x16): vgpr r -> M = r + 8h, N = lane&15.
// ---------------------------------------------------------------------------
template<int MODE, bool OFULL>
__global__ __launch_bounds__(128) void gemm_wmma(GArgs ga) {
  const int lane   = threadIdx.x & 31;
  const int wave   = threadIdx.x >> 5;
  const int oTiles = (ga.O + 15) >> 4;
  const int oT     = blockIdx.y * 4 + wave;
  if (oT >= oTiles) return;                 // wave-uniform
  const int row0 = blockIdx.x << 6;
  const int hh   = lane >> 4;
  const int n    = lane & 15;
  const int oc   = oT * 16 + n;
  const bool oval = OFULL || (oc < ga.O);
  const float* wrow = ga.W + (long long)(oval ? oc : 0) * ga.Kin;

  RowCtx c0 = makeCtx<MODE>(ga, row0 + n);
  RowCtx c1 = makeCtx<MODE>(ga, row0 + 16 + n);
  RowCtx c2 = makeCtx<MODE>(ga, row0 + 32 + n);
  RowCtx c3 = makeCtx<MODE>(ga, row0 + 48 + n);

  v8f acc0 = {}, acc1 = {}, acc2 = {}, acc3 = {};
  const int kSteps = ga.Kin >> 5;
  for (int ks = 0; ks < kSteps; ++ks) {
    const int kbase = ks << 5;
    v16h bf;
    {
      const float* p = wrow + kbase + 8 * hh;
      float4 w0, w1, w2, w3;
      if (oval) { w0 = ld4(p); w1 = ld4(p + 4); w2 = ld4(p + 16); w3 = ld4(p + 20); }
      else      { w0 = w1 = w2 = w3 = make_float4(0.f, 0.f, 0.f, 0.f); }
      pack8(bf, 0, w0, w1);
      pack8(bf, 8, w2, w3);
    }
    v16h a0 = loadA<MODE>(c0, kbase, hh);
    acc0 = __builtin_amdgcn_wmma_f32_16x16x32_f16(false, a0, false, bf, (short)0, acc0, false, false);
    v16h a1 = loadA<MODE>(c1, kbase, hh);
    acc1 = __builtin_amdgcn_wmma_f32_16x16x32_f16(false, a1, false, bf, (short)0, acc1, false, false);
    v16h a2 = loadA<MODE>(c2, kbase, hh);
    acc2 = __builtin_amdgcn_wmma_f32_16x16x32_f16(false, a2, false, bf, (short)0, acc2, false, false);
    v16h a3 = loadA<MODE>(c3, kbase, hh);
    acc3 = __builtin_amdgcn_wmma_f32_16x16x32_f16(false, a3, false, bf, (short)0, acc3, false, false);
  }

  if (oval) {
#pragma unroll
    for (int r = 0; r < 8; ++r) {
      int mb = row0 + r + 8 * hh;
      ga.out[(long long)(mb     ) * ga.O + oc] = acc0[r];
      ga.out[(long long)(mb + 16) * ga.O + oc] = acc1[r];
      ga.out[(long long)(mb + 32) * ga.O + oc] = acc2[r];
      ga.out[(long long)(mb + 48) * ga.O + oc] = acc3[r];
    }
  }
}

// ---------------------------------------------------------------------------
// Stage-1 edge conv: Kin = 6 (C = 3), O = 64. Features live in registers;
// single zero-padded WMMA per tile. grid.x = M/64, 4 waves = 4 O-tiles.
// ---------------------------------------------------------------------------
__global__ __launch_bounds__(128) void gemm_edge6(const float* __restrict__ xt,
                                                  const int* __restrict__ idx,
                                                  const float* __restrict__ W,
                                                  float* __restrict__ out) {
  const int lane = threadIdx.x & 31;
  const int oT   = threadIdx.x >> 5;     // 4 O-tiles cover O=64
  const int row0 = blockIdx.x << 6;
  const int hh   = lane >> 4;
  const int n    = lane & 15;
  const int oc   = oT * 16 + n;

  // B-frag: bf[j] = W[oc][8h + j] (j<8), bf[8+j] = W[oc][16+8h+j], for k < 6, else 0.
  v16h bf;
  const float* wr = W + oc * 6;
#pragma unroll
  for (int j = 0; j < 16; ++j) {
    int k = (j < 8) ? (8 * hh + j) : (16 + 8 * hh + (j - 8));
    bf[j] = (k < 6) ? (_Float16)wr[k] : (_Float16)0.0f;
  }

  v8f acc[4] = {{}, {}, {}, {}};
#pragma unroll
  for (int t = 0; t < 4; ++t) {
    int row = row0 + t * 16 + n;
    int kk = row % KNN_K;
    int pn = row / KNN_K;
    int b  = pn >> 12;
    int nn = pn & (NPTS - 1);
    const float* xb = xt + (long long)b * NPTS * 3;
    int j = idx[(long long)pn * KNN_K + kk];
    float f[6];
#pragma unroll
    for (int c = 0; c < 3; ++c) {
      float xi = xb[nn * 3 + c];
      f[c]     = xb[j * 3 + c] - xi;
      f[3 + c] = xi;
    }
    // af[jj] = A[m][8h + jj] (jj<8), af[8+jj] = A[m][16+8h+jj] -> k>=6 is 0
    v16h af;
#pragma unroll
    for (int jj = 0; jj < 16; ++jj) {
      int k = (jj < 8) ? (8 * hh + jj) : (16 + 8 * hh + (jj - 8));
      af[jj] = (k < 6) ? (_Float16)f[k] : (_Float16)0.0f;
    }
    acc[t] = __builtin_amdgcn_wmma_f32_16x16x32_f16(false, af, false, bf, (short)0, acc[t], false, false);
  }

#pragma unroll
  for (int t = 0; t < 4; ++t)
#pragma unroll
    for (int r = 0; r < 8; ++r)
      out[(long long)(row0 + t * 16 + r + 8 * hh) * 64 + oc] = acc[t][r];
}

// ---------------------------------------------------------------------------
// x:[B,3,N] -> xt:[B,N,3]
// ---------------------------------------------------------------------------
__global__ void transpose_kernel(const float* __restrict__ x, float* __restrict__ xt) {
  int e = blockIdx.x * 256 + threadIdx.x;
  if (e >= BATCH * NPTS * 3) return;
  int c  = e % 3;
  int bn = e / 3;
  int nn = bn & (NPTS - 1);
  int b  = bn >> 12;
  xt[e] = x[((long long)b * 3 + c) * NPTS + nn];
}

// ---------------------------------------------------------------------------
// kNN: one block per (b,i). neg_d into LDS, then 20 argmax extractions
// (tie-break: lowest index, matching lax.top_k).
// ---------------------------------------------------------------------------
template<int C>
__global__ __launch_bounds__(256) void knn_kernel(const float* __restrict__ x,
                                                  int* __restrict__ idx) {
  __shared__ float dist[NPTS];
  __shared__ float rmax[256];
  __shared__ int   rarg[256];
  __shared__ float xi[C];
  const int t = threadIdx.x;
  const int i = blockIdx.x;
  const int b = blockIdx.y;
  const float* xb = x + (long long)b * NPTS * C;

  for (int c = t; c < C; c += 256) xi[c] = xb[(long long)i * C + c];
  __syncthreads();

  float xxi = 0.0f;
#pragma unroll
  for (int c = 0; c < C; ++c) xxi += xi[c] * xi[c];

  for (int j = t; j < NPTS; j += 256) {
    float dot = 0.0f, xxj = 0.0f;
    if (C % 4 == 0) {
      const float4* xj4 = reinterpret_cast<const float4*>(xb + (long long)j * C);
#pragma unroll
      for (int c4 = 0; c4 < C / 4; ++c4) {
        float4 v = xj4[c4];
        dot += xi[4*c4] * v.x + xi[4*c4+1] * v.y + xi[4*c4+2] * v.z + xi[4*c4+3] * v.w;
        xxj += v.x * v.x + v.y * v.y + v.z * v.z + v.w * v.w;
      }
    } else {
      const float* xj = xb + (long long)j * C;
#pragma unroll
      for (int c = 0; c < C; ++c) { float v = xj[c]; dot += xi[c] * v; xxj += v * v; }
    }
    dist[j] = 2.0f * dot - xxi - xxj;
  }
  __syncthreads();

  const float NINF = -__builtin_inff();
  for (int sel = 0; sel < KNN_K; ++sel) {
    float best = NINF; int barg = 0x7fffffff;
    for (int j = t; j < NPTS; j += 256) {
      float v = dist[j];
      if (v > best || (v == best && j < barg)) { best = v; barg = j; }
    }
    rmax[t] = best; rarg[t] = barg;
    __syncthreads();
    for (int s = 128; s > 0; s >>= 1) {
      if (t < s) {
        float v = rmax[t + s]; int a2 = rarg[t + s];
        if (v > rmax[t] || (v == rmax[t] && a2 < rarg[t])) { rmax[t] = v; rarg[t] = a2; }
      }
      __syncthreads();
    }
    if (t == 0) {
      int w = rarg[0];
      idx[((long long)b * NPTS + i) * KNN_K + sel] = w;
      dist[w] = NINF;
    }
    __syncthreads();
  }
}

// ---------------------------------------------------------------------------
// BN stats, deterministic + coalesced. blockDim = max(O,256); o = t % O.
// part: [64 chunks][2*O].
// ---------------------------------------------------------------------------
__global__ void stats_partial(const float* __restrict__ h, float* __restrict__ part,
                              long long M, int O) {
  __shared__ float s1[1024], s2[1024];
  const int t   = threadIdx.x;
  const int BD  = blockDim.x;
  const int o   = t % O;
  const int rl  = t / O;
  const int RPI = BD / O;
  const int chunk = blockIdx.x;        // 64 chunks
  float a = 0.0f, q = 0.0f;
  for (long long r = (long long)chunk * RPI + rl; r < M; r += 64LL * RPI) {
    float v = h[r * O + o];
    a += v; q += v * v;
  }
  s1[t] = a; s2[t] = q;
  __syncthreads();
  if (t < O) {
    for (int g2 = 1; g2 < RPI; ++g2) { a += s1[t + g2 * O]; q += s2[t + g2 * O]; }
    part[(long long)chunk * 2 * O + o]     = a;
    part[(long long)chunk * 2 * O + O + o] = q;
  }
}

__global__ void stats_finalize(const float* __restrict__ part,
                               float* __restrict__ stats, int O) {
  int o = blockIdx.x * 256 + threadIdx.x;
  if (o >= O) return;
  float a = 0.0f, q = 0.0f;
  for (int ch = 0; ch < 64; ++ch) {
    a += part[(long long)ch * 2 * O + o];
    q += part[(long long)ch * 2 * O + O + o];
  }
  stats[o] = a; stats[O + o] = q;
}

__global__ void bn_lrelu_kernel(float* __restrict__ h, const float* __restrict__ stats,
                                const float* __restrict__ g, const float* __restrict__ bia,
                                long long total4, int O, float invM) {
  long long e = (long long)blockIdx.x * 256 + threadIdx.x;
  if (e >= total4) return;
  float4 v = reinterpret_cast<float4*>(h)[e];
  int o = (int)((e * 4) % O);
  float* pv = &v.x;
#pragma unroll
  for (int q = 0; q < 4; ++q) {
    int oq = o + q;
    float mean = stats[oq] * invM;
    float var  = stats[O + oq] * invM - mean * mean;
    float u = (pv[q] - mean) * rsqrtf(var + 1e-5f) * g[oq] + bia[oq];
    pv[q] = (u > 0.0f) ? u : 0.2f * u;
  }
  reinterpret_cast<float4*>(h)[e] = v;
}

// max over k (O=64), float4 vectorized
__global__ void maxk_kernel(const float* __restrict__ h, float* __restrict__ x,
                            long long total4) {
  long long e = (long long)blockIdx.x * 256 + threadIdx.x;
  if (e >= total4) return;
  int o4 = (int)((e * 4) % 64);
  long long p = (e * 4) / 64;
  const float4* hp = reinterpret_cast<const float4*>(h + p * KNN_K * 64 + o4);
  float4 m = hp[0];
#pragma unroll
  for (int kk = 1; kk < KNN_K; ++kk) {
    float4 v = hp[kk * 16];
    m.x = fmaxf(m.x, v.x); m.y = fmaxf(m.y, v.y);
    m.z = fmaxf(m.z, v.z); m.w = fmaxf(m.w, v.w);
  }
  reinterpret_cast<float4*>(x)[e] = m;
}

// global max over N, two-stage deterministic, coalesced over channels
__global__ void gmax_partial(const float* __restrict__ h6, float* __restrict__ pmax) {
  const int c = blockIdx.x * 256 + threadIdx.x;   // 0..1023
  const int b = blockIdx.y;
  const int z = blockIdx.z;                       // 32 chunks of 128 rows
  float m = -__builtin_inff();
  for (int nn = z * 128; nn < z * 128 + 128; ++nn)
    m = fmaxf(m, h6[((long long)b * NPTS + nn) * 1024 + c]);
  pmax[((long long)z * BATCH + b) * 1024 + c] = m;
}

__global__ void gmax_final(const float* __restrict__ pmax, float* __restrict__ gmaxb) {
  const int c = blockIdx.x * 256 + threadIdx.x;
  const int b = blockIdx.y;
  float m = -__builtin_inff();
  for (int z = 0; z < 32; ++z)
    m = fmaxf(m, pmax[((long long)z * BATCH + b) * 1024 + c]);
  gmaxb[b * 1024 + c] = m;
}

// ---------------------------------------------------------------------------
// Host orchestration
// ---------------------------------------------------------------------------
static void launch_gemm(int mode, bool ofull, const GArgs& ga, hipStream_t s) {
  dim3 grid(ga.M / 64, ((ga.O + 15) / 16 + 3) / 4);
  dim3 blk(128);
  switch (mode) {
    case MODE_DENSE:
      if (ofull) gemm_wmma<MODE_DENSE, true><<<grid, blk, 0, s>>>(ga);
      else       gemm_wmma<MODE_DENSE, false><<<grid, blk, 0, s>>>(ga);
      break;
    case MODE_EDGE:    gemm_wmma<MODE_EDGE,    true><<<grid, blk, 0, s>>>(ga); break;
    case MODE_CAT192:  gemm_wmma<MODE_CAT192,  true><<<grid, blk, 0, s>>>(ga); break;
    default:           gemm_wmma<MODE_CAT1216, true><<<grid, blk, 0, s>>>(ga); break;
  }
}

static void bn_stage(float* h, long long M, int O, const float* g, const float* b,
                     float* stats, float* part, hipStream_t s) {
  int BD = (O < 256) ? 256 : O;
  stats_partial<<<dim3(64), BD, 0, s>>>(h, part, M, O);
  stats_finalize<<<dim3((O + 255) / 256), 256, 0, s>>>(part, stats, O);
  long long total4 = M * (long long)O / 4;
  bn_lrelu_kernel<<<dim3((unsigned)((total4 + 255) / 256)), 256, 0, s>>>(
      h, stats, g, b, total4, O, 1.0f / (float)M);
}

extern "C" void kernel_launch(void* const* d_in, const int* in_sizes, int n_in,
                              void* d_out, int out_size, void* d_ws, size_t ws_size,
                              hipStream_t stream) {
  const float* x = (const float*)d_in[0];
  // d_in[1] is k (==20, hardcoded)
  const float* W[9];
  for (int i = 0; i < 9; ++i) W[i] = (const float*)d_in[2 + i];
  const float* g[8]; const float* bb[8];
  for (int i = 0; i < 8; ++i) { g[i] = (const float*)d_in[11 + 2 * i]; bb[i] = (const float*)d_in[12 + 2 * i]; }

  char* ws = (char*)d_ws;
  float* xt    = (float*)(ws + 0);            //   393,216 B  [B,N,3]
  int*   idx   = (int*)  (ws + 393216);       // 2,621,440 B  [B,N,20]
  float* x1    = (float*)(ws + 3014656);      // 8,388,608 B  [B,N,64]
  float* x2    = (float*)(ws + 11403264);     // 8,388,608 B
  float* x3    = (float*)(ws + 19791872);     // 8,388,608 B
  float* gmaxb = (float*)(ws + 28180480);     //    32,768 B  [B,1024]
  float* stats = (float*)(ws + 28213248);     //     8,192 B  [2*1024]
  float* part  = (float*)(ws + 28221440);     //   524,288 B  [64][2*1024]
  float* pmax  = (float*)(ws + 28745728);     // 1,048,576 B  [32][B][1024]
  float* hA    = (float*)(ws + 29794304);     // 167,772,160 B [B*N*20, 64]
  float* hB    = (float*)(ws + 197566464);    // 167,772,160 B
  float* h6 = hB;   // [B*N,1024], aliases hB
  float* h7 = hA;   // [B*N, 512], aliases hA
  float* h8 = hB;   // [B*N, 256], aliases hB

  const long long MEDGE = (long long)BATCH * NPTS * KNN_K;   // 655360
  const long long MPT   = (long long)BATCH * NPTS;           // 32768

  transpose_kernel<<<dim3((BATCH * NPTS * 3 + 255) / 256), 256, 0, stream>>>(x, xt);

  // ---- Stage 1 ----
  knn_kernel<3><<<dim3(NPTS, BATCH), 256, 0, stream>>>(xt, idx);
  gemm_edge6<<<dim3((unsigned)(MEDGE / 64)), 128, 0, stream>>>(xt, idx, W[0], hA);
  bn_stage(hA, MEDGE, 64, g[0], bb[0], stats, part, stream);

  GArgs ga{};
  ga = GArgs{}; ga.A = hA; ga.W = W[1]; ga.out = hB;
  ga.M = (int)MEDGE; ga.Kin = 64; ga.O = 64;
  launch_gemm(MODE_DENSE, true, ga, stream);
  bn_stage(hB, MEDGE, 64, g[1], bb[1], stats, part, stream);
  maxk_kernel<<<dim3((unsigned)((MPT * 16 + 255) / 256)), 256, 0, stream>>>(hB, x1, MPT * 16);

  // ---- Stage 2 ----
  knn_kernel<64><<<dim3(NPTS, BATCH), 256, 0, stream>>>(x1, idx);
  ga = GArgs{}; ga.xsrc = x1; ga.idx = idx; ga.W = W[2]; ga.out = hA;
  ga.M = (int)MEDGE; ga.Kin = 128; ga.O = 64;
  launch_gemm(MODE_EDGE, true, ga, stream);
  bn_stage(hA, MEDGE, 64, g[2], bb[2], stats, part, stream);

  ga = GArgs{}; ga.A = hA; ga.W = W[3]; ga.out = hB;
  ga.M = (int)MEDGE; ga.Kin = 64; ga.O = 64;
  launch_gemm(MODE_DENSE, true, ga, stream);
  bn_stage(hB, MEDGE, 64, g[3], bb[3], stats, part, stream);
  maxk_kernel<<<dim3((unsigned)((MPT * 16 + 255) / 256)), 256, 0, stream>>>(hB, x2, MPT * 16);

  // ---- Stage 3 ----
  knn_kernel<64><<<dim3(NPTS, BATCH), 256, 0, stream>>>(x2, idx);
  ga = GArgs{}; ga.xsrc = x2; ga.idx = idx; ga.W = W[4]; ga.out = hA;
  ga.M = (int)MEDGE; ga.Kin = 128; ga.O = 64;
  launch_gemm(MODE_EDGE, true, ga, stream);
  bn_stage(hA, MEDGE, 64, g[4], bb[4], stats, part, stream);
  maxk_kernel<<<dim3((unsigned)((MPT * 16 + 255) / 256)), 256, 0, stream>>>(hA, x3, MPT * 16);

  // ---- W6: cat192 -> 1024 ----
  ga = GArgs{}; ga.x1 = x1; ga.x2 = x2; ga.x3 = x3; ga.W = W[5]; ga.out = h6;
  ga.M = (int)MPT; ga.Kin = 192; ga.O = 1024;
  launch_gemm(MODE_CAT192, true, ga, stream);
  bn_stage(h6, MPT, 1024, g[5], bb[5], stats, part, stream);

  // ---- global max pool ----
  gmax_partial<<<dim3(4, BATCH, 32), 256, 0, stream>>>(h6, pmax);
  gmax_final<<<dim3(4, BATCH), 256, 0, stream>>>(pmax, gmaxb);

  // ---- W7: cat1216 -> 512 ----
  ga = GArgs{}; ga.gmax = gmaxb; ga.x1 = x1; ga.x2 = x2; ga.x3 = x3;
  ga.W = W[6]; ga.out = h7; ga.M = (int)MPT; ga.Kin = 1216; ga.O = 512;
  launch_gemm(MODE_CAT1216, true, ga, stream);
  bn_stage(h7, MPT, 512, g[6], bb[6], stats, part, stream);

  // ---- W8: 512 -> 256 ----
  ga = GArgs{}; ga.A = h7; ga.W = W[7]; ga.out = h8;
  ga.M = (int)MPT; ga.Kin = 512; ga.O = 256;
  launch_gemm(MODE_DENSE, true, ga, stream);
  bn_stage(h8, MPT, 256, g[7], bb[7], stats, part, stream);

  // ---- W9: 256 -> 9 ----
  ga = GArgs{}; ga.A = h8; ga.W = W[8]; ga.out = (float*)d_out;
  ga.M = (int)MPT; ga.Kin = 256; ga.O = 9;
  launch_gemm(MODE_DENSE, false, ga, stream);
}
